// BicliqueAttentionLayer_50431505989724
// MI455X (gfx1250) — matrix operand order repaired and checked by reading, not compiled, not run on hardware
//
#include <hip/hip_runtime.h>
#include <hip/hip_bf16.h>

// GAT layer for MI455X (gfx1250, wave32).
// Phase 1: h = feat @ (W*mask)^T via v_wmma_f32_16x16x32_f16, W*mask staged in LDS (32KB f16).
// Phase 2: edge scores (coalesced 8-lane gather + shfl_xor reduce), segment-max via
//          order-preserving uint atomicMax, segment-sum via global_atomic_add_f32.
// Phase 3: scatter-aggregate with f32 atomics into d_out, in-place ReLU.
// All heavy traffic (h = 12.8MB, accumulators) is L2-resident (192MB L2).

typedef __attribute__((ext_vector_type(16))) _Float16 v16h;
typedef __attribute__((ext_vector_type(8)))  float    v8f;

#define N_NODES 50000
#define N_EDGES 800000
#define IN_DIM  256
#define OUT_DIM 64
#define TILES   (N_NODES / 16)   // 3125 exact 16-row tiles

// ---------------- Phase 0: zero-init output + reduction buffers ----------------
__global__ __launch_bounds__(256) void gat_init(float* __restrict__ out,
                                                unsigned* __restrict__ smax_key,
                                                float* __restrict__ denom) {
  int i = blockIdx.x * blockDim.x + threadIdx.x;
  if (i < N_NODES * OUT_DIM) out[i] = 0.0f;
  if (i < N_NODES) { smax_key[i] = 0u; denom[i] = 0.0f; }
}

// ---------------- Phase 1: masked GEMM via WMMA ----------------
// One wave per 16-node tile; wave computes 16x64 output (4 accumulators).
__global__ __launch_bounds__(128) void gat_gemm_wmma(const float* __restrict__ feat,
                                                     const float* __restrict__ mask,
                                                     const float* __restrict__ W,
                                                     float* __restrict__ h) {
  __shared__ __align__(32) _Float16 Wlds[OUT_DIM * IN_DIM];  // Wm[n][k] = W[n][k]*mask[k], 32KB
  const int tid = threadIdx.x;
  for (int i = tid; i < OUT_DIM * IN_DIM; i += 128) {
    int k = i & (IN_DIM - 1);
    Wlds[i] = (_Float16)(W[i] * mask[k]);
  }
  __syncthreads();

  const int wave = tid >> 5;
  const int lane = tid & 31;
  const int hl   = lane & 15;   // position within half-wave
  const int hi   = lane >> 4;   // which half-wave
  const int tile = blockIdx.x * 4 + wave;
  if (tile >= TILES) return;

  const int rowBase = tile * 16;
  const float* arow = feat + (size_t)(rowBase + hl) * IN_DIM;

  v8f acc0 = {}, acc1 = {}, acc2 = {}, acc3 = {};

#pragma unroll
  for (int kk = 0; kk < IN_DIM; kk += 32) {
    // A fragment (16x32 f16): lane holds K = kk + hi*16 + {0..15}.
    const float4* p0 = (const float4*)(arow + kk + hi * 16);
    const float4* p1 = (const float4*)(arow + kk + hi * 16 + 8);
    float4 f0 = p0[0], f1 = p0[1], f2 = p1[0], f3 = p1[1];
    float tmp[16] = {f0.x, f0.y, f0.z, f0.w, f1.x, f1.y, f1.z, f1.w,
                     f2.x, f2.y, f2.z, f2.w, f3.x, f3.y, f3.z, f3.w};
    v16h a;
#pragma unroll
    for (int j = 0; j < 16; ++j) a[j] = (_Float16)tmp[j];

    // B fragments (32x16 f16): lane's column n = nb*16+hl, K = kk + hi*16 + {0..15},
    // read as 16 contiguous halfs from the LDS-resident Wm row (32B aligned -> ds_load_b128 x2).
    const v16h* b0 = (const v16h*)&Wlds[(0 * 16 + hl) * IN_DIM + kk + hi * 16];
    const v16h* b1 = (const v16h*)&Wlds[(1 * 16 + hl) * IN_DIM + kk + hi * 16];
    const v16h* b2 = (const v16h*)&Wlds[(2 * 16 + hl) * IN_DIM + kk + hi * 16];
    const v16h* b3 = (const v16h*)&Wlds[(3 * 16 + hl) * IN_DIM + kk + hi * 16];

    acc0 = __builtin_amdgcn_wmma_f32_16x16x32_f16(false, a, false, *b0, (short)0, acc0, false, false);
    acc1 = __builtin_amdgcn_wmma_f32_16x16x32_f16(false, a, false, *b1, (short)0, acc1, false, false);
    acc2 = __builtin_amdgcn_wmma_f32_16x16x32_f16(false, a, false, *b2, (short)0, acc2, false, false);
    acc3 = __builtin_amdgcn_wmma_f32_16x16x32_f16(false, a, false, *b3, (short)0, acc3, false, false);
  }

  // C/D layout: VGPR r -> M = r + hi*8, N = hl (+ nb*16).
  float* hrow = h + (size_t)(rowBase + hi * 8) * OUT_DIM + hl;
#pragma unroll
  for (int r = 0; r < 8; ++r) {
    hrow[(size_t)r * OUT_DIM + 0]  = acc0[r];
    hrow[(size_t)r * OUT_DIM + 16] = acc1[r];
    hrow[(size_t)r * OUT_DIM + 32] = acc2[r];
    hrow[(size_t)r * OUT_DIM + 48] = acc3[r];
  }
}

// Order-preserving float->uint key (monotone for atomicMax).
__device__ __forceinline__ unsigned f2key(float f) {
  unsigned x = __float_as_uint(f);
  return (x & 0x80000000u) ? ~x : (x | 0x80000000u);
}
__device__ __forceinline__ float key2f(unsigned key) {
  unsigned x = (key & 0x80000000u) ? (key ^ 0x80000000u) : ~key;
  return __uint_as_float(x);
}

// ---------------- Phase 2a: edge scores + segment max ----------------
__global__ __launch_bounds__(256) void gat_scores(const float* __restrict__ h,
                                                  const float* __restrict__ attn,
                                                  const int* __restrict__ src,
                                                  const int* __restrict__ dst,
                                                  float* __restrict__ score,
                                                  unsigned* __restrict__ smax_key) {
  int t = blockIdx.x * blockDim.x + threadIdx.x;
  int e = t >> 3, lane = t & 7;              // 8 lanes per edge: 256B coalesced row read
  if (e >= N_EDGES) return;
  int s = src[e];
  const float4* hp = (const float4*)(h + (size_t)s * OUT_DIM) + lane * 2;
  const float4* ap = (const float4*)attn + lane * 2;
  float4 v0 = hp[0], v1 = hp[1];
  float4 a0 = ap[0], a1 = ap[1];
  float p = v0.x * a0.x + v0.y * a0.y + v0.z * a0.z + v0.w * a0.w +
            v1.x * a1.x + v1.y * a1.y + v1.z * a1.z + v1.w * a1.w;
  p += __shfl_xor(p, 1, 32);
  p += __shfl_xor(p, 2, 32);
  p += __shfl_xor(p, 4, 32);
  if (lane == 0) {
    float sc = p > 0.0f ? p : 0.01f * p;     // leaky_relu
    score[e] = sc;
    atomicMax(&smax_key[dst[e]], f2key(sc));
  }
}

// ---------------- Phase 2b: exp(score - max) + segment sum ----------------
__global__ __launch_bounds__(256) void gat_expsum(const float* __restrict__ score,
                                                  const int* __restrict__ dst,
                                                  const unsigned* __restrict__ smax_key,
                                                  float* __restrict__ exbuf,
                                                  float* __restrict__ denom) {
  int e = blockIdx.x * blockDim.x + threadIdx.x;
  if (e >= N_EDGES) return;
  int d = dst[e];
  float ex = __expf(score[e] - key2f(smax_key[d]));
  exbuf[e] = ex;
  atomicAdd(&denom[d], ex);
}

// ---------------- Phase 3a: scatter-aggregate h[src]*alpha into out[dst] ----------------
__global__ __launch_bounds__(256) void gat_aggregate(const float* __restrict__ h,
                                                     const int* __restrict__ src,
                                                     const int* __restrict__ dst,
                                                     const float* __restrict__ exbuf,
                                                     const float* __restrict__ denom,
                                                     float* __restrict__ out) {
  int t = blockIdx.x * blockDim.x + threadIdx.x;
  int e = t >> 3, lane = t & 7;
  if (e >= N_EDGES) return;
  int s = src[e], d = dst[e];
  float alpha = exbuf[e] / denom[d];
  const float4* hp = (const float4*)(h + (size_t)s * OUT_DIM) + lane * 2;
  float4 v0 = hp[0], v1 = hp[1];
  float* op = out + (size_t)d * OUT_DIM + lane * 8;
  atomicAdd(op + 0, v0.x * alpha);
  atomicAdd(op + 1, v0.y * alpha);
  atomicAdd(op + 2, v0.z * alpha);
  atomicAdd(op + 3, v0.w * alpha);
  atomicAdd(op + 4, v1.x * alpha);
  atomicAdd(op + 5, v1.y * alpha);
  atomicAdd(op + 6, v1.z * alpha);
  atomicAdd(op + 7, v1.w * alpha);
}

// ---------------- Phase 3b: in-place ReLU ----------------
__global__ __launch_bounds__(256) void gat_relu(float* __restrict__ out) {
  int i = blockIdx.x * blockDim.x + threadIdx.x;
  if (i < N_NODES * OUT_DIM) out[i] = fmaxf(out[i], 0.0f);
}

extern "C" void kernel_launch(void* const* d_in, const int* in_sizes, int n_in,
                              void* d_out, int out_size, void* d_ws, size_t ws_size,
                              hipStream_t stream) {
  const float* feat = (const float*)d_in[0];   // [50000, 256]
  const float* mask = (const float*)d_in[1];   // [256]
  const float* W    = (const float*)d_in[2];   // [64, 256]
  const float* attn = (const float*)d_in[3];   // [64]
  const int*   src  = (const int*)d_in[4];     // [800000]
  const int*   dst  = (const int*)d_in[5];     // [800000], sorted
  float* out = (float*)d_out;                  // [50000, 64]

  char* ws = (char*)d_ws;
  float*    h     = (float*)(ws);                 // 12,800,000 B
  float*    score = (float*)(ws + 12800000);      //  3,200,000 B
  float*    exbuf = (float*)(ws + 16000000);      //  3,200,000 B
  unsigned* smax  = (unsigned*)(ws + 19200000);   //    200,000 B
  float*    denom = (float*)(ws + 19400000);      //    200,000 B

  // init out/smax/denom (harness poisons buffers; re-zero every call for determinism)
  gat_init<<<(N_NODES * OUT_DIM + 255) / 256, 256, 0, stream>>>(out, smax, denom);
  // h = feat @ (W*mask)^T  (WMMA, 4 waves/block, 1 tile/wave)
  gat_gemm_wmma<<<(TILES + 3) / 4, 128, 0, stream>>>(feat, mask, W, h);
  // scores + segment max (8 lanes/edge)
  gat_scores<<<(N_EDGES * 8 + 255) / 256, 256, 0, stream>>>(h, attn, src, dst, score, smax);
  // exp + segment sum
  gat_expsum<<<(N_EDGES + 255) / 256, 256, 0, stream>>>(score, dst, smax, exbuf, denom);
  // weighted scatter-sum
  gat_aggregate<<<(N_EDGES * 8 + 255) / 256, 256, 0, stream>>>(h, src, dst, exbuf, denom, out);
  // relu
  gat_relu<<<(N_NODES * OUT_DIM + 255) / 256, 256, 0, stream>>>(out);
}
